// GraphConvolution_16913581211891
// MI455X (gfx1250) — compile-verified, matrix operand order checked by reading, not run on hardware
//
#include <hip/hip_runtime.h>
#include <hip/hip_bf16.h>
#include <math.h>

// ---------------------------------------------------------------------------
// ACM-GCN layer for MI455X (gfx1250, wave32, WMMA + TDM).
//   N=8192, IN_F=512, OUT_F=256.
// Dominant GEMM: S = f16(g) @ XW  (8192x8192 @ 8192x512, f32 accum), then
//   adj_low@XW = rinv * (S + XW)   (per-row epilogue; adj never materialized)
//   O_low = relu(that), O_high = relu(XW_high - that).
// B tiles move Global->LDS via the Tensor Data Mover (TENSORcnt), A tiles are
// register-prefetched f32->f16 converted; B LDS is double buffered.
// ---------------------------------------------------------------------------

typedef __attribute__((ext_vector_type(16))) _Float16 v16h;
typedef __attribute__((ext_vector_type(8)))  _Float16 v8h;
typedef __attribute__((ext_vector_type(4)))  _Float16 v4h;
typedef __attribute__((ext_vector_type(8)))  float    v8f;

#define NN   8192
#define INF  512
#define OUTF 256
#define KD   512   // cols of XW = [low|high]
#define CD   768   // cols of WT = [low|high|mlp]

#if defined(__has_builtin)
#if __has_builtin(__builtin_amdgcn_tensor_load_to_lds) && \
    __has_builtin(__builtin_amdgcn_s_wait_tensorcnt)
#define USE_TDM 1
#endif
#if __has_builtin(__builtin_amdgcn_sched_barrier)
#define SCHED_SPLIT() __builtin_amdgcn_sched_barrier(0)
#endif
#endif
#ifndef USE_TDM
#define USE_TDM 0
#endif
#ifndef SCHED_SPLIT
#define SCHED_SPLIT()
#endif

#if USE_TDM
typedef __attribute__((ext_vector_type(4))) unsigned int v4u;
typedef __attribute__((ext_vector_type(8))) int          v8i;
typedef __attribute__((ext_vector_type(4))) int          v4i;

// 2D f16 tile: Global -> LDS, tile_d0 contiguous elements per row, tile_d1
// rows, row stride d0_stride elements; LDS gets pad_amount DWORDs inserted
// every pad_interval DWORDs (encodings per D# group1). Issue from one wave.
// 6-arg builtin: (u32x4 g0, i32x8 g1, i32x4 g2, i32x4 g3, i32x8, i32 cpol)
__device__ __forceinline__ void tdm_load_2d_f16(
    unsigned lds_off, const void* gaddr,
    unsigned tensor_d0, unsigned tensor_d1, unsigned d0_stride,
    unsigned tile_d0, unsigned tile_d1,
    unsigned pad_interval_enc, unsigned pad_amount_enc) {
  unsigned long long ga = (unsigned long long)gaddr;
  v4u g0;
  g0[0] = 1u;                                   // count=1, user descriptor
  g0[1] = lds_off;                              // LDS byte address
  g0[2] = (unsigned)ga;                         // global_addr[31:0]
  g0[3] = (unsigned)((ga >> 32) & 0x1FFFFFFu)   // global_addr[56:32]
          | (2u << 30);                         // type = 2 ("image")
  v8i g1;
  g1[0] = (int)((1u << 16)                      // data_size = 1 -> 2 bytes
                | (1u << 20)                    // pad_enable
                | (pad_interval_enc << 22)
                | (pad_amount_enc << 25));
  g1[1] = (int)(tensor_d0 << 16);               // tensor_dim0[15:0] @bits63:48
  g1[2] = (int)((tensor_d0 >> 16) | (tensor_d1 << 16));
  g1[3] = (int)((tensor_d1 >> 16) | (tile_d0 << 16));
  g1[4] = (int)(tile_d1 & 0xFFFFu);             // tile_dim1; tile_dim2 = 0
  g1[5] = (int)d0_stride;                       // tensor_dim0_stride[31:0]
  g1[6] = 0;                                    // stride[47:32] | dim1_stride lo
  g1[7] = 0;
  v4i z4 = {0, 0, 0, 0};
  v8i z8 = {0, 0, 0, 0, 0, 0, 0, 0};
  __builtin_amdgcn_tensor_load_to_lds(g0, g1, z4, z4, z8, 0);
}
#endif

// ---------------------------------------------------------------------------
// Kernel W: pack {W_low,W_high,W_mlp} ([512,256] f32 row-major) into
// WT[768][512] f16, K-contiguous (WT[c][k] = W_sel[k][c_local]).
// ---------------------------------------------------------------------------
__global__ __launch_bounds__(256) void wconv(const float* __restrict__ wl,
                                             const float* __restrict__ wh,
                                             const float* __restrict__ wm,
                                             _Float16* __restrict__ WT) {
  int idx = blockIdx.x * 256 + threadIdx.x;     // 768*512 total
  if (idx >= CD * INF) return;
  int c = idx >> 9;
  int k = idx & 511;
  float v;
  if (c < 256)      v = wl[k * OUTF + c];
  else if (c < 512) v = wh[k * OUTF + (c - 256)];
  else              v = wm[k * OUTF + (c - 512)];
  WT[idx] = (_Float16)v;
}

// ---------------------------------------------------------------------------
// Kernel A: per-row reductions, one wave32 per row.
// ---------------------------------------------------------------------------
__global__ __launch_bounds__(256) void rowsums(const float* __restrict__ g,
                                               const float* __restrict__ feats,
                                               float* __restrict__ rinv,
                                               float* __restrict__ finv) {
  const int w = threadIdx.x >> 5, l = threadIdx.x & 31;
  const int row = blockIdx.x * 8 + w;
  const float* gr = g + (size_t)row * NN;
  float s = 0.f;
  for (int k = l * 4; k < NN; k += 128) {
    float4 v = *(const float4*)(gr + k);
    s += v.x + v.y + v.z + v.w;
  }
  for (int off = 16; off; off >>= 1) s += __shfl_xor(s, off, 32);
  if (l == 0) rinv[row] = 1.0f / (1.0f + s);

  const float* fr = feats + (size_t)row * INF;
  float sf = 0.f;
  for (int k = l * 4; k < INF; k += 128) {
    float4 v = *(const float4*)(fr + k);
    sf += v.x + v.y + v.z + v.w;
  }
  for (int off = 16; off; off >>= 1) sf += __shfl_xor(sf, off, 32);
  if (l == 0) finv[row] = (sf == 0.f) ? 0.f : 1.0f / sf;
}

// ---------------------------------------------------------------------------
// Kernel B: XW-GEMM via WMMA. S = feats_f16 @ WT^T -> [8192,768].
// Epilogue: val=finv[row]*S; c<512 -> XWT[c][row] f16; c>=512 -> O_mlp=relu.
// 512 thr = 16 waves: 4 row-tiles x 4 col-groups(192 = 12 tiles).
// WT tile staged by TDM (pad 4 DW every 16 DW -> 40-half LDS stride).
// ---------------------------------------------------------------------------
__global__ __launch_bounds__(512, 1)
void xw_gemm(const float* __restrict__ feats, const _Float16* __restrict__ WT,
             const float* __restrict__ finv, _Float16* __restrict__ XWT,
             float* __restrict__ Omlp) {
  __shared__ __align__(16) _Float16 As[64 * 40];   // 64 rows x 32 K (pad 40)
  __shared__ __align__(16) _Float16 Bs[768 * 40];  // 768 cols x 32 K (pad 40)

  const int t = threadIdx.x;
  const int w = t >> 5, l = t & 31;
  const int rg = w & 3, cg = w >> 2;
  const int lr = l & 15, lh = l >> 4;
  const int rowbase = blockIdx.x * 64;

  v8f acc[12] = {};

  for (int k0 = 0; k0 < INF; k0 += 32) {
#if USE_TDM
    if (t < 32)
      tdm_load_2d_f16((unsigned)(size_t)&Bs[0], WT + k0,
                      /*tensor_d0=*/INF, /*tensor_d1=*/CD, /*stride=*/INF,
                      /*tile_d0=*/32, /*tile_d1=*/CD,
                      /*pad_interval: 16 DW=*/3, /*pad_amount: 4 DW=*/3);
#else
    for (int c = t; c < CD; c += 512) {
      const _Float16* src = WT + (size_t)c * INF + k0;
#pragma unroll
      for (int j = 0; j < 4; ++j)
        *(v8h*)&Bs[c * 40 + j * 8] = *(const v8h*)(src + j * 8);
    }
#endif
    // stage A: 64x32 f32 -> f16 LDS
    {
      int row = t >> 3, k4 = (t & 7) * 4;
      float4 f = *(const float4*)(feats + (size_t)(rowbase + row) * INF + k0 + k4);
      v4h h; h[0] = (_Float16)f.x; h[1] = (_Float16)f.y;
      h[2] = (_Float16)f.z; h[3] = (_Float16)f.w;
      *(v4h*)&As[row * 40 + k4] = h;
    }
#if USE_TDM
    if (t < 32) __builtin_amdgcn_s_wait_tensorcnt(0);
#endif
    __syncthreads();

    const _Float16* ap = &As[(rg * 16 + lr) * 40 + lh * 8];
    v8h a0 = *(const v8h*)ap;
    v8h a1 = *(const v8h*)(ap + 16);
    v16h a = __builtin_shufflevector(a0, a1, 0, 1, 2, 3, 4, 5, 6, 7,
                                     8, 9, 10, 11, 12, 13, 14, 15);
    v16h bf[12];
#pragma unroll
    for (int ct = 0; ct < 12; ++ct) {
      const _Float16* bp = &Bs[(cg * 192 + ct * 16 + lr) * 40 + lh * 8];
      v8h b0 = *(const v8h*)bp;
      v8h b1 = *(const v8h*)(bp + 16);
      bf[ct] = __builtin_shufflevector(b0, b1, 0, 1, 2, 3, 4, 5, 6, 7,
                                       8, 9, 10, 11, 12, 13, 14, 15);
    }
    SCHED_SPLIT();   // batch all ds_loads before the WMMA chain
#pragma unroll
    for (int ct = 0; ct < 12; ++ct)
      acc[ct] = __builtin_amdgcn_wmma_f32_16x16x32_f16(
          false, a, false, bf[ct], (short)0, acc[ct], false, false);
    __syncthreads();
  }

#pragma unroll
  for (int v = 0; v < 8; ++v) {
    const int row = rowbase + rg * 16 + v + 8 * lh;
    const float f = finv[row];
#pragma unroll
    for (int ct = 0; ct < 12; ++ct) {
      const int c = cg * 192 + ct * 16 + lr;
      const float val = f * acc[ct][v];
      if (c < KD) XWT[(size_t)c * NN + row] = (_Float16)val;
      else        Omlp[(size_t)row * OUTF + (c - KD)] = fmaxf(val, 0.f);
    }
  }
}

// ---------------------------------------------------------------------------
// Kernel C: the big GEMM. 128 WGs x 512 thr (16 waves): 64-row stripe x all
// 512 cols -> g read ONCE from HBM. B tile (XWT, L2-resident) double-buffered
// via TDM; A tile register-prefetched (f32->f16) and stored mid-iteration.
// ---------------------------------------------------------------------------
__global__ __launch_bounds__(512, 1)
void main_gemm(const float* __restrict__ g, const _Float16* __restrict__ XWT,
               const float* __restrict__ rinv, float* __restrict__ Olow,
               float* __restrict__ Ohigh) {
  __shared__ __align__(16) _Float16 As[64 * 72];       // 64 rows x 64 K (pad 72)
  __shared__ __align__(16) _Float16 Bs[2][512 * 72];   // 512 cols x 64 K (pad 72)

  const int t = threadIdx.x;
  const int w = t >> 5, l = t & 31;
  const int rg = w & 3, cg = w >> 2;                   // 4 row-tiles x 4 col-grps
  const int lr = l & 15, lh = l >> 4;
  const int rowbase = blockIdx.x * 64;

  const int arow = t >> 3, akk = (t & 7) * 8;          // A staging assignment
  const float* asrc = g + (size_t)(rowbase + arow) * NN + akk;

  v8f acc[8] = {};

  // ---- preamble: stage tile 0 ----
#if USE_TDM
  if (t < 32)
    tdm_load_2d_f16((unsigned)(size_t)&Bs[0][0], XWT,
                    /*tensor_d0=*/NN, /*tensor_d1=*/KD, /*stride=*/NN,
                    /*tile_d0=*/64, /*tile_d1=*/KD,
                    /*pad_interval: 32 DW=*/4, /*pad_amount: 4 DW=*/3);
#else
  {
    const _Float16* src = XWT + (size_t)t * NN;
#pragma unroll
    for (int j = 0; j < 8; ++j)
      *(v8h*)&Bs[0][t * 72 + j * 8] = *(const v8h*)(src + j * 8);
  }
#endif
  {
    float4 f0 = *(const float4*)asrc;
    float4 f1 = *(const float4*)(asrc + 4);
    v8h h;
    h[0] = (_Float16)f0.x; h[1] = (_Float16)f0.y;
    h[2] = (_Float16)f0.z; h[3] = (_Float16)f0.w;
    h[4] = (_Float16)f1.x; h[5] = (_Float16)f1.y;
    h[6] = (_Float16)f1.z; h[7] = (_Float16)f1.w;
    *(v8h*)&As[arow * 72 + akk] = h;
  }
#if USE_TDM
  if (t < 32) __builtin_amdgcn_s_wait_tensorcnt(0);
#endif
  __syncthreads();

  int buf = 0;
  for (int k0 = 0; k0 < NN; k0 += 64) {
    const bool next = (k0 + 64) < NN;

    // prefetch next A tile into registers (wait deferred past the As fragment reads)
    float4 f0 = {}, f1 = {};
    if (next) {
      f0 = *(const float4*)(asrc + k0 + 64);
      f1 = *(const float4*)(asrc + k0 + 68);
    }
#if USE_TDM
    if (next && t < 32)
      tdm_load_2d_f16((unsigned)(size_t)&Bs[buf ^ 1][0], XWT + k0 + 64,
                      NN, KD, NN, 64, KD, 4, 3);
#endif
    // pull this tile's A fragments (both K-substeps) into registers
    v16h afrag[2];
#pragma unroll
    for (int s = 0; s < 2; ++s) {
      const _Float16* ap = &As[(rg * 16 + lr) * 72 + s * 32 + lh * 8];
      v8h a0 = *(const v8h*)ap;
      v8h a1 = *(const v8h*)(ap + 16);
      afrag[s] = __builtin_shufflevector(a0, a1, 0, 1, 2, 3, 4, 5, 6, 7,
                                         8, 9, 10, 11, 12, 13, 14, 15);
    }
    __syncthreads();   // everyone done reading As -> safe to overwrite

    if (next) {        // convert + store next A tile (overlaps with WMMA below)
      v8h h;
      h[0] = (_Float16)f0.x; h[1] = (_Float16)f0.y;
      h[2] = (_Float16)f0.z; h[3] = (_Float16)f0.w;
      h[4] = (_Float16)f1.x; h[5] = (_Float16)f1.y;
      h[6] = (_Float16)f1.z; h[7] = (_Float16)f1.w;
      *(v8h*)&As[arow * 72 + akk] = h;
    }
#if !USE_TDM
    if (next) {
      const _Float16* src = XWT + (size_t)t * NN + k0 + 64;
#pragma unroll
      for (int j = 0; j < 8; ++j)
        *(v8h*)&Bs[buf ^ 1][t * 72 + j * 8] = *(const v8h*)(src + j * 8);
    }
#endif
    // compute on (As, Bs[buf])
#pragma unroll
    for (int s = 0; s < 2; ++s) {
      v16h bfrag[8];
#pragma unroll
      for (int ct = 0; ct < 8; ++ct) {
        const _Float16* bp =
            &Bs[buf][(cg * 128 + ct * 16 + lr) * 72 + s * 32 + lh * 8];
        v8h b0 = *(const v8h*)bp;
        v8h b1 = *(const v8h*)(bp + 16);
        bfrag[ct] = __builtin_shufflevector(b0, b1, 0, 1, 2, 3, 4, 5, 6, 7,
                                            8, 9, 10, 11, 12, 13, 14, 15);
      }
      SCHED_SPLIT();   // batch the 16 ds_loads before the 8-WMMA chain
#pragma unroll
      for (int ct = 0; ct < 8; ++ct)
        acc[ct] = __builtin_amdgcn_wmma_f32_16x16x32_f16(
            false, afrag[s], false, bfrag[ct], (short)0, acc[ct], false, false);
    }
#if USE_TDM
    if (next && t < 32) __builtin_amdgcn_s_wait_tensorcnt(0);
#endif
    __syncthreads();   // next tile (A stores + B TDM) visible to all
    buf ^= 1;
  }

  // epilogue: adj_low@XW = rinv*(S + XW); low -> relu(t); high -> relu(xw - t)
#pragma unroll
  for (int v = 0; v < 8; ++v) {
    const int row = rowbase + rg * 16 + v + 8 * lh;
    const float rin = rinv[row];
#pragma unroll
    for (int ct = 0; ct < 8; ++ct) {
      const int c = cg * 128 + ct * 16 + lr;
      const float xw = (float)XWT[(size_t)c * NN + row];
      const float tlo = rin * (acc[ct][v] + xw);
      if (c < OUTF) Olow[(size_t)row * OUTF + c] = fmaxf(tlo, 0.f);
      else          Ohigh[(size_t)row * OUTF + (c - OUTF)] = fmaxf(xw - tlo, 0.f);
    }
  }
}

// ---------------------------------------------------------------------------
// Kernel D: attention3 + blend. One 256-thread block per row.
// ---------------------------------------------------------------------------
__global__ __launch_bounds__(256)
void attn(const float* __restrict__ Olow, const float* __restrict__ Ohigh,
          const float* __restrict__ Omlp, const float* __restrict__ avl,
          const float* __restrict__ avh, const float* __restrict__ avm,
          const float* __restrict__ attv, float* __restrict__ out) {
  __shared__ float red[3][8];
  __shared__ float attw[3];
  const int row = blockIdx.x, c = threadIdx.x;
  const size_t base = (size_t)row * OUTF + c;
  const float ol = Olow[base], oh = Ohigh[base], om = Omlp[base];
  float p0 = ol * avl[c], p1 = oh * avh[c], p2 = om * avm[c];
  for (int off = 16; off; off >>= 1) {
    p0 += __shfl_xor(p0, off, 32);
    p1 += __shfl_xor(p1, off, 32);
    p2 += __shfl_xor(p2, off, 32);
  }
  if ((c & 31) == 0) {
    red[0][c >> 5] = p0; red[1][c >> 5] = p1; red[2][c >> 5] = p2;
  }
  __syncthreads();
  if (c == 0) {
    float d0 = 0.f, d1 = 0.f, d2 = 0.f;
#pragma unroll
    for (int i = 0; i < 8; ++i) { d0 += red[0][i]; d1 += red[1][i]; d2 += red[2][i]; }
    const float s0 = 1.f / (1.f + expf(-d0));
    const float s1 = 1.f / (1.f + expf(-d1));
    const float s2 = 1.f / (1.f + expf(-d2));
    float lg[3];
#pragma unroll
    for (int j = 0; j < 3; ++j)
      lg[j] = (s0 * attv[0 * 3 + j] + s1 * attv[1 * 3 + j] + s2 * attv[2 * 3 + j]) * (1.f / 3.f);
    const float m = fmaxf(lg[0], fmaxf(lg[1], lg[2]));
    const float e0 = expf(lg[0] - m), e1 = expf(lg[1] - m), e2 = expf(lg[2] - m);
    const float z = 1.f / (e0 + e1 + e2);
    attw[0] = e0 * z; attw[1] = e1 * z; attw[2] = e2 * z;
  }
  __syncthreads();
  out[base] = 3.f * (attw[0] * ol + attw[1] * oh + attw[2] * om);
}

// ---------------------------------------------------------------------------
extern "C" void kernel_launch(void* const* d_in, const int* in_sizes, int n_in,
                              void* d_out, int out_size, void* d_ws, size_t ws_size,
                              hipStream_t stream) {
  const float* g     = (const float*)d_in[0];
  const float* feats = (const float*)d_in[1];
  // d_in[2] = ep (unused by the forward math)
  const float* wl    = (const float*)d_in[3];
  const float* wh    = (const float*)d_in[4];
  const float* wm    = (const float*)d_in[5];
  const float* avl   = (const float*)d_in[6];
  const float* avh   = (const float*)d_in[7];
  const float* avm   = (const float*)d_in[8];
  const float* attv  = (const float*)d_in[9];
  float* out = (float*)d_out;

  // workspace carve-up (~32.8 MiB total)
  char* ws = (char*)d_ws;
  float*    rinv = (float*)ws;                                    // 8192 f32
  float*    finv = rinv + NN;                                     // 8192 f32
  _Float16* WT   = (_Float16*)(finv + NN);                        // 768*512 f16
  _Float16* XWT  = (_Float16*)((char*)WT + (size_t)CD * INF * 2); // 512*8192 f16
  float*    Olow = (float*)((char*)XWT + (size_t)KD * NN * 2);    // 8192*256 f32
  float*    Ohigh = Olow + (size_t)NN * OUTF;
  float*    Omlp  = Ohigh + (size_t)NN * OUTF;

  wconv<<<(CD * INF + 255) / 256, 256, 0, stream>>>(wl, wh, wm, WT);
  rowsums<<<NN / 8, 256, 0, stream>>>(g, feats, rinv, finv);
  xw_gemm<<<NN / 64, 512, 0, stream>>>(feats, WT, finv, XWT, Omlp);
  main_gemm<<<NN / 64, 512, 0, stream>>>(g, XWT, rinv, Olow, Ohigh);
  attn<<<NN, 256, 0, stream>>>(Olow, Ohigh, Omlp, avl, avh, avm, attv, out);
}
